// SparseAttention_32908039421977
// MI455X (gfx1250) — matrix operand order repaired and checked by reading, not compiled, not run on hardware
//
#include <hip/hip_runtime.h>
#include <hip/hip_bf16.h>
#include <stdint.h>

typedef __attribute__((ext_vector_type(16))) __bf16 v16bf;
typedef __attribute__((ext_vector_type(8)))  float  v8f;

union FragAB {
  v16bf v;
  unsigned u[8];
  uint4 q[2];
  unsigned short h[16];
};

__device__ __forceinline__ unsigned short f32_to_bf16_rne(float f) {
  unsigned u = __float_as_uint(f);
  u += ((u >> 16) & 1u) + 0x7FFFu;
  return (unsigned short)(u >> 16);
}

// Async global -> LDS copy of 16B, CDNA5 path (ASYNCcnt tracked).
// inst_offset is added to BOTH the LDS and the global address (ISA 10.x async).
#define ASYNC_CP16(ldsaddr, gaddr, imm_off)                                    \
  asm volatile("global_load_async_to_lds_b128 %0, %1, off offset:" #imm_off    \
               ::"v"(ldsaddr), "v"(gaddr) : "memory")

#define ASYNC_WAIT0() asm volatile("s_wait_asynccnt 0x0" ::: "memory")

__device__ __forceinline__ unsigned lds_addr32(const void* p) {
  return (unsigned)(size_t)p;  // flat->LDS: hardware uses addr[31:0]
}

// ---------------------------------------------------------------------------
// f32 -> bf16 conversion, 4 elements / thread
// ---------------------------------------------------------------------------
__global__ __launch_bounds__(256) void cvt_f32_bf16_k(const float* __restrict__ in,
                                                      unsigned short* __restrict__ out) {
  long i = ((long)blockIdx.x * blockDim.x + threadIdx.x) * 4;
  float4 f = *(const float4*)(in + i);
  uint2 p;
  p.x = (unsigned)f32_to_bf16_rne(f.x) | ((unsigned)f32_to_bf16_rne(f.y) << 16);
  p.y = (unsigned)f32_to_bf16_rne(f.z) | ((unsigned)f32_to_bf16_rne(f.w) << 16);
  *(uint2*)(out + i) = p;
}

// ---------------------------------------------------------------------------
// V transpose: Vt[b][c][n] = qkv[b][n][1536 + c]   (bf16), 32x32 LDS tiles
// ---------------------------------------------------------------------------
__global__ __launch_bounds__(256)
void transpose_v_k(const unsigned short* __restrict__ qkv,
                   unsigned short* __restrict__ Vt) {
  __shared__ unsigned short tile[32][33];
  const int b = blockIdx.z;
  const int n0 = blockIdx.x * 32, c0 = blockIdx.y * 32;
  const int tc = threadIdx.x & 31, tr = threadIdx.x >> 5;  // 32 x 8
#pragma unroll
  for (int i = 0; i < 4; ++i) {
    int n = tr + i * 8;
    tile[n][tc] = qkv[((long)b * 2048 + n0 + n) * 2304 + 1536 + c0 + tc];
  }
  __syncthreads();
#pragma unroll
  for (int i = 0; i < 4; ++i) {
    int c = tr + i * 8;
    Vt[((long)b * 768 + c0 + c) * 2048 + n0 + tc] = tile[tc][c];
  }
}

// ---------------------------------------------------------------------------
// Tiled WMMA GEMM (NT): C[M,N] = scale * A[M,K] * B^T, B stored [N,K].
// Block tile 128x128, K step 32, 256 threads = 8 waves, wave tile 32x64.
// Double-buffered LDS tiles filled by GLOBAL_LOAD_ASYNC_TO_LDS_B128;
// one barrier per K step, DMA of tile kt+1 overlaps WMMA on tile kt.
// ---------------------------------------------------------------------------
template <bool OUT_BF16>
__global__ __launch_bounds__(256)
void gemm_wmma_k(const unsigned short* __restrict__ A, long lda, long strideA,
                 const unsigned short* __restrict__ B, long ldb, long strideB,
                 void* __restrict__ Cv, long ldc, long strideC,
                 int Kdim, float scale) {
  __shared__ unsigned short As[2][128][40];  // +8-half pad, 10240 B per buffer
  __shared__ unsigned short Bs[2][128][40];

  const int nb = blockIdx.x, mb = blockIdx.y, zb = blockIdx.z;
  A += (long)zb * strideA + (long)mb * 128 * lda;
  B += (long)zb * strideB;
  const long n_base = (long)nb * 128;

  const int t = threadIdx.x;
  const int lane = t & 31;
  const int w   = t >> 5;
  const int wm  = w & 3;    // m sub-tile: wm*32
  const int wn  = w >> 2;   // n sub-tile: wn*64
  const int h   = lane >> 4;
  const int l16 = lane & 15;

  // per-thread staging slice: row = t/2 (0..127), col = (t&1)*16 halves
  const int srow = t >> 1;
  const int scol = (t & 1) << 4;
  const unsigned short* aSrc = A + (long)srow * lda + scol;
  const unsigned short* bSrc = B + (n_base + srow) * ldb + scol;
  const unsigned aDst0 = lds_addr32(&As[0][srow][scol]);
  const unsigned bDst0 = lds_addr32(&Bs[0][srow][scol]);
  const unsigned bufStep = 128u * 40u * 2u;  // 10240 B

  auto issue = [&](int kt, int p) {
    const unsigned short* ga = aSrc + (long)kt * 32;
    const unsigned short* gb = bSrc + (long)kt * 32;
    unsigned ad = aDst0 + (unsigned)p * bufStep;
    unsigned bd = bDst0 + (unsigned)p * bufStep;
    ASYNC_CP16(ad, ga, 0);
    ASYNC_CP16(ad, ga, 16);
    ASYNC_CP16(bd, gb, 0);
    ASYNC_CP16(bd, gb, 16);
  };

  v8f acc[2][4];
#pragma unroll
  for (int i = 0; i < 2; ++i)
#pragma unroll
    for (int j = 0; j < 4; ++j)
#pragma unroll
      for (int r = 0; r < 8; ++r) acc[i][j][r] = 0.f;

  const int ksteps = Kdim >> 5;

  issue(0, 0);
  ASYNC_WAIT0();
  __syncthreads();  // buffer 0 ready for everyone

  for (int kt = 0; kt < ksteps; ++kt) {
    const int p = kt & 1;
    if (kt + 1 < ksteps) issue(kt + 1, p ^ 1);  // overwrites buffer read at kt-1 (safe: barrier)

    // ---- gather fragments per ISA 7.12.2 layouts, compute ----
    FragAB af[2], bfr[4];
#pragma unroll
    for (int i = 0; i < 2; ++i) {
      const unsigned short* Arow = &As[p][wm * 32 + i * 16 + l16][0];
#pragma unroll
      for (int g = 0; g < 8; ++g) {
        int kk = ((g < 4) ? 0 : 16) + h * 8 + (g & 3) * 2;
        af[i].u[g] = *(const unsigned*)(Arow + kk);
      }
    }
#pragma unroll
    for (int j = 0; j < 4; ++j) {
      const unsigned short* Brow = &Bs[p][wn * 64 + j * 16 + l16][h * 8];
      bfr[j].q[0] = *(const uint4*)Brow;
      bfr[j].q[1] = *(const uint4*)(Brow + 16);
    }
#pragma unroll
    for (int i = 0; i < 2; ++i)
#pragma unroll
      for (int j = 0; j < 4; ++j)
        acc[i][j] = __builtin_amdgcn_wmma_f32_16x16x32_bf16(
            false, af[i].v, false, bfr[j].v, (short)0, acc[i][j], false, false);

    ASYNC_WAIT0();    // tile kt+1 landed (DMA overlapped the WMMAs above)
    __syncthreads();  // all waves' LDS reads of buffer p done; next buffer ready
  }

  // ---- epilogue: D layout -> VGPR r holds row (h*8 + r), lane%16 = col ----
  if (OUT_BF16) {
    unsigned short* Cp = (unsigned short*)Cv + (long)zb * strideC;
#pragma unroll
    for (int i = 0; i < 2; ++i)
#pragma unroll
      for (int j = 0; j < 4; ++j) {
        long col = n_base + wn * 64 + j * 16 + l16;
#pragma unroll
        for (int r = 0; r < 8; ++r) {
          long rrow = (long)mb * 128 + wm * 32 + i * 16 + h * 8 + r;
          Cp[rrow * ldc + col] = f32_to_bf16_rne(acc[i][j][r] * scale);
        }
      }
  } else {
    float* Cp = (float*)Cv + (long)zb * strideC;
#pragma unroll
    for (int i = 0; i < 2; ++i)
#pragma unroll
      for (int j = 0; j < 4; ++j) {
        long col = n_base + wn * 64 + j * 16 + l16;
#pragma unroll
        for (int r = 0; r < 8; ++r) {
          long rrow = (long)mb * 128 + wm * 32 + i * 16 + h * 8 + r;
          Cp[rrow * ldc + col] = acc[i][j][r] * scale;
        }
      }
  }
}

// ---------------------------------------------------------------------------
// Per-row exact top-k threshold via 4x8-bit radix select on order-preserving
// uint keys, then masked softmax. One 256-thread block per row of 2048.
// ---------------------------------------------------------------------------
__global__ __launch_bounds__(256)
void topk_softmax_k(const float* __restrict__ attn, unsigned short* __restrict__ P,
                    int kkeep) {
  __shared__ unsigned hist[256];
  __shared__ unsigned sufx[256];
  __shared__ float    red[256];
  __shared__ unsigned selb;
  __shared__ unsigned selrem;

  const int t = threadIdx.x;
  const long row = blockIdx.x;
  const float* src = attn + row * 2048;

  float v[8];
  unsigned key[8];
  float4 f0 = *(const float4*)(src + t * 8);
  float4 f1 = *(const float4*)(src + t * 8 + 4);
  v[0] = f0.x; v[1] = f0.y; v[2] = f0.z; v[3] = f0.w;
  v[4] = f1.x; v[5] = f1.y; v[6] = f1.z; v[7] = f1.w;
#pragma unroll
  for (int i = 0; i < 8; ++i) {
    unsigned u = __float_as_uint(v[i]);
    key[i] = u ^ ((u >> 31) ? 0xFFFFFFFFu : 0x80000000u);  // monotonic key
  }

  unsigned prefix = 0, done = 0, remaining = (unsigned)kkeep;
#pragma unroll
  for (int pass = 0; pass < 4; ++pass) {
    const int shift = 24 - pass * 8;
    hist[t] = 0;
    __syncthreads();
#pragma unroll
    for (int i = 0; i < 8; ++i)
      if ((key[i] & done) == prefix) atomicAdd(&hist[(key[i] >> shift) & 255u], 1u);
    __syncthreads();
    sufx[t] = hist[t];
    __syncthreads();
    for (int off = 1; off < 256; off <<= 1) {  // suffix sums
      unsigned add = (t + off < 256) ? sufx[t + off] : 0u;
      __syncthreads();
      sufx[t] += add;
      __syncthreads();
    }
    unsigned mine = sufx[t];
    unsigned nxt = (t == 255) ? 0u : sufx[t + 1];
    if (mine >= remaining && nxt < remaining) {
      selb = (unsigned)t;
      selrem = remaining - nxt;
    }
    __syncthreads();
    prefix |= selb << shift;
    done |= 0xFFu << shift;
    remaining = selrem;
    __syncthreads();
  }
  const unsigned T = prefix;  // key of the k-th largest element

  float m = -INFINITY;
#pragma unroll
  for (int i = 0; i < 8; ++i) m = fmaxf(m, v[i]);
  red[t] = m;
  __syncthreads();
  for (int off = 128; off; off >>= 1) {
    if (t < off) red[t] = fmaxf(red[t], red[t + off]);
    __syncthreads();
  }
  float rowmax = red[0];
  __syncthreads();

  float e[8];
  float s = 0.f;
#pragma unroll
  for (int i = 0; i < 8; ++i) {
    bool keep = key[i] >= T;
    e[i] = keep ? expf(v[i] - rowmax) : 0.f;
    s += e[i];
  }
  red[t] = s;
  __syncthreads();
  for (int off = 128; off; off >>= 1) {
    if (t < off) red[t] += red[t + off];
    __syncthreads();
  }
  float inv = 1.f / red[0];

  unsigned short oh[8];
#pragma unroll
  for (int i = 0; i < 8; ++i) oh[i] = f32_to_bf16_rne(e[i] * inv);
  uint4 q;
  q.x = (unsigned)oh[0] | ((unsigned)oh[1] << 16);
  q.y = (unsigned)oh[2] | ((unsigned)oh[3] << 16);
  q.z = (unsigned)oh[4] | ((unsigned)oh[5] << 16);
  q.w = (unsigned)oh[6] | ((unsigned)oh[7] << 16);
  *(uint4*)(P + row * 2048 + t * 8) = q;
}

// ---------------------------------------------------------------------------
extern "C" void kernel_launch(void* const* d_in, const int* in_sizes, int n_in,
                              void* d_out, int out_size, void* d_ws, size_t ws_size,
                              hipStream_t stream) {
  (void)in_sizes; (void)n_in; (void)out_size; (void)ws_size;
  const float* x = (const float*)d_in[0];
  const float* W = (const float*)d_in[1];
  float* out = (float*)d_out;

  const int Bn = 8, N = 2048, C = 768;
  const long xN    = (long)Bn * N * C;       // 12,582,912
  const long wN    = (long)3 * C * C;        //  1,769,472
  const long qkvN  = (long)Bn * N * 3 * C;   // 37,748,736
  const long attnN = (long)Bn * N * N;       // 33,554,432
  const long vN    = (long)Bn * C * N;       // 12,582,912

  char* ws = (char*)d_ws;
  unsigned short* xb  = (unsigned short*)ws; ws += xN * 2;
  unsigned short* wb  = (unsigned short*)ws; ws += wN * 2;
  unsigned short* qkv = (unsigned short*)ws; ws += qkvN * 2;
  float*          att = (float*)ws;          ws += attnN * 4;
  unsigned short* P   = (unsigned short*)ws; ws += attnN * 2;
  unsigned short* Vt  = (unsigned short*)ws; ws += vN * 2;

  // 1) convert inputs to bf16
  cvt_f32_bf16_k<<<(unsigned)(xN / 1024), 256, 0, stream>>>(x, xb);
  cvt_f32_bf16_k<<<(unsigned)(wN / 1024), 256, 0, stream>>>(W, wb);

  // 2) qkv[M=16384, 3C] = x[M, C] * W^T   (W stored [3C, C], K-contiguous)
  gemm_wmma_k<true><<<dim3(3 * C / 128, Bn * N / 128, 1), 256, 0, stream>>>(
      xb, C, 0, wb, C, 0, (void*)qkv, 3 * C, 0, C, 1.0f);

  // 3) Vt[b][c][n] = V[b][n][c]
  transpose_v_k<<<dim3(N / 32, C / 32, Bn), 256, 0, stream>>>(qkv, Vt);

  // 4) attn[b, N, N] = Q * K^T / sqrt(C)
  const float sc = 1.0f / sqrtf((float)C);
  gemm_wmma_k<false><<<dim3(N / 128, N / 128, Bn), 256, 0, stream>>>(
      qkv, 3 * C, (long)N * 3 * C,
      qkv + C, 3 * C, (long)N * 3 * C,
      (void*)att, N, (long)N * N, C, sc);

  // 5) exact top-k (k = int(0.8*2048) = 1638) threshold + softmax -> bf16 P
  topk_softmax_k<<<Bn * N, 256, 0, stream>>>(att, P, 1638);

  // 6) out[b, N, C] = P[b, N, N] * Vt[b, C, N]^T
  gemm_wmma_k<false><<<dim3(C / 128, N / 128, Bn), 256, 0, stream>>>(
      P, N, (long)N * N,
      Vt, N, (long)C * N,
      (void*)out, C, (long)N * C, N, 1.0f);
}
